// RNN_8555574853747
// MI455X (gfx1250) — compile-verified, matrix operand order checked by reading, not run on hardware
//
#include <hip/hip_runtime.h>

#define DEV static __device__ __forceinline__

// ---------- types ----------
typedef __bf16 bf16_t;
typedef bf16_t bf16x16 __attribute__((ext_vector_type(16)));
typedef float  f32x8   __attribute__((ext_vector_type(8)));
typedef float  f32x4   __attribute__((ext_vector_type(4)));
typedef unsigned short u16x8 __attribute__((ext_vector_type(8)));

// Problem dims
#define BB   32
#define TT   1024
#define DIN  512
#define DH   1024
#define DOUT 512

// ---------- helpers ----------
DEV bf16_t f2bf(float f) {
  unsigned u = __builtin_bit_cast(unsigned, f);
  unsigned r = (u + 0x7FFFu + ((u >> 16) & 1u)) >> 16;   // round-to-nearest-even
  unsigned short s = (unsigned short)r;
  return __builtin_bit_cast(bf16_t, s);
}
DEV unsigned short f2bfu(float f) {
  return __builtin_bit_cast(unsigned short, f2bf(f));
}

// bf16 fragment load: lane (0..31) -> row = lane&15, K-half = lane>>4.
// Same mapping used for A and B operands => consistent K permutation.
DEV bf16x16 frag_from_bf16(const unsigned short* base, int ld) {
  int lane = threadIdx.x & 31;
  return *(const bf16x16*)(base + (size_t)(lane & 15) * ld + ((lane >> 4) << 4));
}
DEV const unsigned short* lane_ptr_bf16(const unsigned short* base, int ld) {
  int lane = threadIdx.x & 31;
  return base + (size_t)(lane & 15) * ld + ((lane >> 4) << 4);
}

DEV f32x8 wmma_bf16(bf16x16 a, bf16x16 b, f32x8 c) {
  return __builtin_amdgcn_wmma_f32_16x16x32_bf16(
      /*neg_a=*/false, a, /*neg_b=*/false, b,
      /*c_mod=*/(short)0, c, /*reuse_a=*/false, /*reuse_b=*/false);
}

DEV f32x8 zero8() { f32x8 z;
#pragma unroll
  for (int e = 0; e < 8; ++e) z[e] = 0.0f; return z; }

// ============================================================
// Kernel 0: elementwise f32 -> bf16 (RNE), 8 elems/thread.
// ============================================================
__global__ __launch_bounds__(256)
void k_cvt(const float* __restrict__ src, unsigned short* __restrict__ dst,
           int n8) {
  int i = blockIdx.x * blockDim.x + threadIdx.x;
  if (i >= n8) return;
  f32x4 a = ((const f32x4*)src)[2 * i];
  f32x4 b = ((const f32x4*)src)[2 * i + 1];
  u16x8 o;
#pragma unroll
  for (int e = 0; e < 4; ++e) { o[e] = f2bfu(a[e]); o[4 + e] = f2bfu(b[e]); }
  ((u16x8*)dst)[i] = o;
}

// ============================================================
// Shared GEMM body: wave tile 64x64 (4x4 frags), block 256 thr (8 waves 2x4),
// block tile 128M x 256N. Per K-step: 8 frag loads (16 b128) -> 16 WMMAs.
// ============================================================

// ============================================================
// Kernel 1: xp[T][B][H] = x[B*T][DIN] @ Wx^T + (Wx_b + Wh_b)
// ============================================================
__global__ __launch_bounds__(256, 1)
void k_xproj(const unsigned short* __restrict__ X,   // [32768][512] bf16
             const unsigned short* __restrict__ Wx,  // [1024][512] bf16
             const float* __restrict__ bx, const float* __restrict__ bh,
             float* __restrict__ XP) {
  const int K = DIN;
  int w = threadIdx.x >> 5, lane = threadIdx.x & 31;
  int wm = w >> 2, wn = w & 3;
  int m0 = blockIdx.x * 128 + wm * 64;
  int n0 = blockIdx.y * 256 + wn * 64;

  f32x8 acc[4][4];
#pragma unroll
  for (int i = 0; i < 4; ++i)
#pragma unroll
    for (int j = 0; j < 4; ++j) acc[i][j] = zero8();

  for (int k0 = 0; k0 < K; k0 += 32) {
    if (k0 + 32 < K) {
      __builtin_prefetch(lane_ptr_bf16(X + (size_t)m0 * K + k0 + 32, K), 0, 1);
      __builtin_prefetch(lane_ptr_bf16(Wx + (size_t)n0 * K + k0 + 32, K), 0, 1);
    }
    bf16x16 a[4], b[4];
#pragma unroll
    for (int i = 0; i < 4; ++i)
      a[i] = frag_from_bf16(X + (size_t)(m0 + i * 16) * K + k0, K);
#pragma unroll
    for (int j = 0; j < 4; ++j)
      b[j] = frag_from_bf16(Wx + (size_t)(n0 + j * 16) * K + k0, K);
#pragma unroll
    for (int i = 0; i < 4; ++i)
#pragma unroll
      for (int j = 0; j < 4; ++j) acc[i][j] = wmma_bf16(a[i], b[j], acc[i][j]);
  }

#pragma unroll
  for (int j = 0; j < 4; ++j) {
    int n = n0 + j * 16 + (lane & 15);
    float bias = bx[n] + bh[n];
#pragma unroll
    for (int i = 0; i < 4; ++i)
#pragma unroll
      for (int v = 0; v < 8; ++v) {
        int m = m0 + i * 16 + v + ((lane >> 4) << 3);  // row in x = b*T + t
        int t = m & (TT - 1), bb = m >> 10;
        XP[((size_t)t * BB + bb) * DH + n] = acc[i][j][v] + bias;
      }
  }
}

// ============================================================
// Kernel 2: persistent recurrent scan.
// 32 WGs x 128 thr (4 waves). WG wg owns H-columns [wg*32, wg*32+32).
// Wh slice (32x1024 bf16 = 64KB) preloaded into LDS in WMMA B-frag layout
// => Wh touches HBM exactly once for all 1024 steps.
// Per step: h_{t+1} = sigmoid(h_t @ Wh^T + xp_t); grid-sync via atomic.
// ============================================================
__global__ __launch_bounds__(128, 1)
void k_scan(const unsigned short* __restrict__ Wh,  // [1024][1024] bf16
            const float* __restrict__ XP,           // [T][B][H] f32
            unsigned short* __restrict__ Hall,      // [T+1][B][H] bf16
            unsigned* __restrict__ bar) {
  __shared__ unsigned short whs[2 * 32 * 32 * 16];  // [ntl][kblk][lane][16] = 64KB
  int lane = threadIdx.x & 31;
  int w = threadIdx.x >> 5;     // 0..3
  int nbase = blockIdx.x * 32;

  // preload + swizzle Wh slice into LDS (once)
  for (int c = w; c < 64; c += 4) {                 // c = ntl*32 + kblk
    int ntl = c >> 5, kblk = c & 31;
    bf16x16 f = frag_from_bf16(
        Wh + (size_t)(nbase + ntl * 16) * DH + kblk * 32, DH);
    *(bf16x16*)&whs[((size_t)c * 32 + lane) * 16] = f;
  }
  __syncthreads();

  int mtile = w & 1, ntl = w >> 1;
  int m0 = mtile * 16;
  int nglob = nbase + ntl * 16;

  for (int t = 0; t < TT; ++t) {
    const unsigned short* hprev = Hall + (size_t)t * BB * DH;
    f32x8 c = zero8();
#pragma unroll 4
    for (int kblk = 0; kblk < 32; ++kblk) {
      bf16x16 a = frag_from_bf16(hprev + (size_t)m0 * DH + kblk * 32, DH);
      bf16x16 b = *(const bf16x16*)&whs[((size_t)((ntl << 5) | kblk) * 32 + lane) * 16];
      c = wmma_bf16(a, b, c);
    }
    // epilogue: + xp, sigmoid, store bf16 h_{t+1}
    unsigned short* hnext = Hall + (size_t)(t + 1) * BB * DH;
    const float* xp = XP + (size_t)t * BB * DH;
    int ncol = nglob + (lane & 15);
#pragma unroll
    for (int v = 0; v < 8; ++v) {
      int m = m0 + v + ((lane >> 4) << 3);
      float s = c[v] + xp[(size_t)m * DH + ncol];
      s = 1.0f / (1.0f + __expf(-s));
      hnext[(size_t)m * DH + ncol] = f2bfu(s);
    }
    // ---- grid barrier (monotonic counter, 32 WGs) ----
    __threadfence();
    __syncthreads();
    if (threadIdx.x == 0) {
      __hip_atomic_fetch_add(bar, 1u, __ATOMIC_RELEASE, __HIP_MEMORY_SCOPE_AGENT);
      unsigned target = 32u * (unsigned)(t + 1);
      while (__hip_atomic_load(bar, __ATOMIC_ACQUIRE, __HIP_MEMORY_SCOPE_AGENT) < target)
        __builtin_amdgcn_s_sleep(1);
    }
    __syncthreads();
    __threadfence();
  }
}

// ============================================================
// Kernel 3: y[b][t][o] = h_all[t*B+b][:] @ Wy^T + Wy_b  (A,B bf16)
// ============================================================
__global__ __launch_bounds__(256, 1)
void k_yproj(const unsigned short* __restrict__ Hs,  // [32768][1024] bf16
             const unsigned short* __restrict__ Wy,  // [512][1024] bf16
             const float* __restrict__ by, float* __restrict__ Y) {
  const int K = DH;
  int w = threadIdx.x >> 5, lane = threadIdx.x & 31;
  int wm = w >> 2, wn = w & 3;
  int m0 = blockIdx.x * 128 + wm * 64;
  int n0 = blockIdx.y * 256 + wn * 64;

  f32x8 acc[4][4];
#pragma unroll
  for (int i = 0; i < 4; ++i)
#pragma unroll
    for (int j = 0; j < 4; ++j) acc[i][j] = zero8();

  for (int k0 = 0; k0 < K; k0 += 32) {
    if (k0 + 32 < K) {
      __builtin_prefetch(lane_ptr_bf16(Hs + (size_t)m0 * K + k0 + 32, K), 0, 1);
      __builtin_prefetch(lane_ptr_bf16(Wy + (size_t)n0 * K + k0 + 32, K), 0, 1);
    }
    bf16x16 a[4], b[4];
#pragma unroll
    for (int i = 0; i < 4; ++i)
      a[i] = frag_from_bf16(Hs + (size_t)(m0 + i * 16) * K + k0, K);
#pragma unroll
    for (int j = 0; j < 4; ++j)
      b[j] = frag_from_bf16(Wy + (size_t)(n0 + j * 16) * K + k0, K);
#pragma unroll
    for (int i = 0; i < 4; ++i)
#pragma unroll
      for (int j = 0; j < 4; ++j) acc[i][j] = wmma_bf16(a[i], b[j], acc[i][j]);
  }

#pragma unroll
  for (int j = 0; j < 4; ++j) {
    int n = n0 + j * 16 + (lane & 15);
    float bias = by[n];
#pragma unroll
    for (int i = 0; i < 4; ++i)
#pragma unroll
      for (int v = 0; v < 8; ++v) {
        int m = m0 + i * 16 + v + ((lane >> 4) << 3);  // row = t*B + b
        int t = m >> 5, bb = m & 31;
        Y[((size_t)bb * TT + t) * DOUT + n] = acc[i][j][v] + bias;
      }
  }
}

// ============================================================
extern "C" void kernel_launch(void* const* d_in, const int* in_sizes, int n_in,
                              void* d_out, int out_size, void* d_ws, size_t ws_size,
                              hipStream_t stream) {
  (void)in_sizes; (void)n_in; (void)out_size; (void)ws_size;
  const float* x   = (const float*)d_in[0];
  const float* Whw = (const float*)d_in[1];
  const float* Whb = (const float*)d_in[2];
  const float* Wxw = (const float*)d_in[3];
  const float* Wxb = (const float*)d_in[4];
  const float* Wyw = (const float*)d_in[5];
  const float* Wyb = (const float*)d_in[6];
  float* y = (float*)d_out;

  // ---- workspace layout (all chunks 256B-aligned) ----
  char* ws = (char*)d_ws;
  size_t off = 0;
  unsigned* bar = (unsigned*)(ws + off);            off += 256;
  unsigned short* hall = (unsigned short*)(ws + off);
  off += (size_t)(TT + 1) * BB * DH * 2;            // 67,174,400
  float* xp = (float*)(ws + off);
  off += (size_t)TT * BB * DH * 4;                  // 134,217,728
  unsigned short* xbf  = (unsigned short*)(ws + off); off += (size_t)BB * TT * DIN * 2;
  unsigned short* whbf = (unsigned short*)(ws + off); off += (size_t)DH * DH * 2;
  unsigned short* wxbf = (unsigned short*)(ws + off); off += (size_t)DH * DIN * 2;
  unsigned short* wybf = (unsigned short*)(ws + off); off += (size_t)DOUT * DH * 2;

  hipMemsetAsync(bar, 0, 256, stream);
  hipMemsetAsync(hall, 0, (size_t)BB * DH * 2, stream);   // h_0 = 0 slab

  // one-shot f32 -> bf16 conversions (bandwidth-bound)
  {
    int nx = BB * TT * DIN / 8;   // 2,097,152
    int nh = DH * DH / 8;         //   131,072
    int nw = DH * DIN / 8;        //    65,536
    int no = DOUT * DH / 8;       //    65,536
    k_cvt<<<dim3(nx / 256), dim3(256), 0, stream>>>(x,   xbf,  nx);
    k_cvt<<<dim3(nh / 256), dim3(256), 0, stream>>>(Whw, whbf, nh);
    k_cvt<<<dim3(nw / 256), dim3(256), 0, stream>>>(Wxw, wxbf, nw);
    k_cvt<<<dim3(no / 256), dim3(256), 0, stream>>>(Wyw, wybf, no);
  }

  k_xproj<<<dim3(256, 4), dim3(256), 0, stream>>>(xbf, wxbf, Wxb, Whb, xp);
  k_scan <<<dim3(32),     dim3(128), 0, stream>>>(whbf, xp, hall, bar);
  k_yproj<<<dim3(256, 2), dim3(256), 0, stream>>>(hall + (size_t)BB * DH, wybf, Wyb, y);
}
